// GraphGCN_24464133718825
// MI455X (gfx1250) — matrix operand order repaired and checked by reading, not compiled, not run on hardware
//
#include <hip/hip_runtime.h>

typedef __attribute__((ext_vector_type(16))) _Float16 v16h;
typedef __attribute__((ext_vector_type(8)))  float    v8f;

#define SEQ_LEN   25
#define HID       100
#define EMBED     300
#define NTWEETS   40000
#define NUSERS    30000
#define NNODES    70000
#define NEDGES    200000
#define BATCH     4000
#define G3        300   // 3*HID

#define LDS_AS __attribute__((address_space(3)))

// ---- gfx1250 async global->LDS path via inline asm (doc-sanctioned route;
//      bypasses the clang builtin whose pointer AS we cannot spell in HIP).
//      Sync staging fallback for the host pass / other targets. ----
#if defined(__gfx1250__)
#define USE_ASYNC_LDS 1
__device__ __forceinline__ unsigned lds_off_u32(const void* p)
{
    return (unsigned)(unsigned long long)(LDS_AS const void*)p;
}
__device__ __forceinline__ void async_load_b128(const void* g, const void* lds)
{
    unsigned long long ga = (unsigned long long)g;
    unsigned           la = lds_off_u32(lds);
    asm volatile("global_load_async_to_lds_b128 %0, %1, off"
                 :: "v"(la), "v"(ga) : "memory");
}
__device__ __forceinline__ void async_wait0()
{
    asm volatile("s_wait_asynccnt 0" ::: "memory");
}
#endif

// ---------------------------------------------------------------------------
// WMMA GEMM:  C[M,N] = A @ W + bias
//   A: direct f32 [M, lda] (gather==0), or rows gathered from the embedding
//      table via tokens[m*SEQ+t] (gather==1, row length == K).
//   W: f32 [K, N] row-major.  bias: f32[N] or nullptr.
// Block = 128 threads (4 waves). Block tile: 32 M-rows x (8*16=128) N-cols.
//   - A tile (2 x 16x32, f32) staged in LDS via global_load_async_to_lds_b128
//   - each wave owns 2 N-subtiles of B (f16, transposed [n][k]) and
//     issues 4 x v_wmma_f32_16x16x32_f16 per K-step.
// grid.x = ceil(M/32), grid.y = ceil(ceil(N/16)/8)
// ---------------------------------------------------------------------------
__global__ __launch_bounds__(128)
void wmma_gemm_kernel(const float* __restrict__ A, int lda,
                      const int* __restrict__ tokens, int t,
                      const float* __restrict__ emb,
                      const float* __restrict__ W,
                      const float* __restrict__ bias,
                      float* __restrict__ C,
                      int M, int N, int K, int gather)
{
    __shared__ __align__(16) float    sAf[2][16][32];   // 2 M-subtiles, f32
    __shared__ __align__(16) _Float16 sB[8][16 * 32];   // 8 N-subtiles, [n][k]

    const int tid   = threadIdx.x;
    const int wave  = tid >> 5;
    const int lane  = tid & 31;
    const int mtile = blockIdx.x;                 // 32 rows
    const int nb    = blockIdx.y * 8 + wave * 2;  // first of 2 N-subtiles
    const bool mPartial = (mtile * 32 + 32 > M);

    v8f acc00 = {}, acc01 = {}, acc10 = {}, acc11 = {};

    for (int kk = 0; kk < K; kk += 32) {
        const bool kPartial = (kk + 32 > K);

#if defined(USE_ASYNC_LDS)
        // pre-zero A tile when padding is needed (DS stores ordered before
        // the async writes by the barrier below)
        if (kPartial || mPartial) {
            for (int i = tid; i < 1024; i += 128) ((float*)sAf)[i] = 0.0f;
            __syncthreads();
        }
        // async global->LDS: 32 rows x 8 quads of 4 floats (16B, aligned)
        for (int slot = tid; slot < 256; slot += 128) {
            int r = slot >> 3, c = slot & 7;
            int m = mtile * 32 + r;
            int k = kk + c * 4;                   // K%4==0 -> quad-exact mask
            if (m < M && k < K) {
                const float* g;
                if (gather) {
                    int tok = tokens[(size_t)m * SEQ_LEN + t];
                    g = emb + (size_t)tok * K + k;
                } else {
                    g = A + (size_t)m * lda + k;
                }
                async_load_b128(g, &sAf[r >> 4][r & 15][c * 4]);
            }
        }
#else
        // sync staging fallback
        for (int i = tid; i < 1024; i += 128) {
            int r = i >> 5, c = i & 31;
            int m = mtile * 32 + r;
            int k = kk + c;
            float v = 0.0f;
            if (m < M && k < K) {
                if (gather) {
                    int tok = tokens[(size_t)m * SEQ_LEN + t];
                    v = emb[(size_t)tok * K + k];
                } else {
                    v = A[(size_t)m * lda + k];
                }
            }
            sAf[r >> 4][r & 15][c] = v;
        }
#endif
        // stage this wave's two B subtiles (f32 -> f16, transposed [n][k])
        for (int s = 0; s < 2; ++s) {
            int n0s = (nb + s) * 16;
            for (int i = lane; i < 512; i += 32) {
                int n = i >> 5, c = i & 31;
                int k = kk + c, nn = n0s + n;
                float v = 0.0f;
                if (k < K && nn < N) v = W[(size_t)k * N + nn];
                sB[wave * 2 + s][n * 32 + c] = (_Float16)v;
            }
        }
#if defined(USE_ASYNC_LDS)
        async_wait0();   // drain this wave's ASYNCcnt before publishing tile
#endif
        __syncthreads();

        // ---- fragments (CDNA5 wave32 layouts) ----
        // A 16x32 f16: lane 0-15 -> M=lane, K=0..7 & 16..23 ; lanes 16-31 +8
        const int mr = lane & 15;
        const int kb = (lane >> 4) * 8;
        v16h av0, av1;
#pragma unroll
        for (int j = 0; j < 8; ++j) {
            av0[j]     = (_Float16)sAf[0][mr][kb + j];
            av0[j + 8] = (_Float16)sAf[0][mr][kb + 16 + j];
            av1[j]     = (_Float16)sAf[1][mr][kb + j];
            av1[j + 8] = (_Float16)sAf[1][mr][kb + 16 + j];
        }
        // B 32x16 f16: lane 0-15 -> N=lane, K=0..7 & 16..23 ; lanes 16-31 +8
        const int nc  = lane & 15;
        const int kb2 = (lane >> 4) * 8;
        v16h bv0, bv1;
#pragma unroll
        for (int j = 0; j < 16; ++j) {
            bv0[j] = sB[wave * 2 + 0][nc * 32 + kb2 + j + (j >= 8 ? 8 : 0)];
            bv1[j] = sB[wave * 2 + 1][nc * 32 + kb2 + j + (j >= 8 ? 8 : 0)];
        }

        acc00 = __builtin_amdgcn_wmma_f32_16x16x32_f16(
                    false, av0, false, bv0, (short)0, acc00, false, false);
        acc01 = __builtin_amdgcn_wmma_f32_16x16x32_f16(
                    false, av0, false, bv1, (short)0, acc01, false, false);
        acc10 = __builtin_amdgcn_wmma_f32_16x16x32_f16(
                    false, av1, false, bv0, (short)0, acc10, false, false);
        acc11 = __builtin_amdgcn_wmma_f32_16x16x32_f16(
                    false, av1, false, bv1, (short)0, acc11, false, false);
        __syncthreads();
    }

    // ---- store: VGPR r, lane group -> M = r + 8*(lane>>4), N = lane&15 ----
    const int nc  = lane & 15;
    const int hi  = (lane >> 4) << 3;
    const int nn0 = nb * 16 + nc;
    const int nn1 = (nb + 1) * 16 + nc;
    const float b0 = (bias && nn0 < N) ? bias[nn0] : 0.0f;
    const float b1 = (bias && nn1 < N) ? bias[nn1] : 0.0f;
#pragma unroll
    for (int r = 0; r < 8; ++r) {
        int m0 = mtile * 32 + r + hi;
        int m1 = m0 + 16;
        if (m0 < M && nn0 < N) C[(size_t)m0 * N + nn0] = acc00[r] + b0;
        if (m0 < M && nn1 < N) C[(size_t)m0 * N + nn1] = acc01[r] + b1;
        if (m1 < M && nn0 < N) C[(size_t)m1 * N + nn0] = acc10[r] + b0;
        if (m1 < M && nn1 < N) C[(size_t)m1 * N + nn1] = acc11[r] + b1;
    }
}

// ---------------------------------------------------------------------------
// User encoder MLP: 9 -> 100 (relu) -> 100.  One block per user.
// ---------------------------------------------------------------------------
__global__ __launch_bounds__(128)
void user_mlp_kernel(const float* __restrict__ uf,
                     const float* __restrict__ fc1w, const float* __restrict__ fc1b,
                     const float* __restrict__ fc2w, const float* __restrict__ fc2b,
                     float* __restrict__ out)
{
    __shared__ float sf[9];
    __shared__ float sh1[100];
    const int u = blockIdx.x;
    const int j = threadIdx.x;
    if (j < 9) sf[j] = uf[(size_t)u * 9 + j];
    __syncthreads();
    if (j < 100) {
        float a = fc1b[j];
#pragma unroll
        for (int i = 0; i < 9; ++i) a += sf[i] * fc1w[i * 100 + j];
        sh1[j] = a > 0.0f ? a : 0.0f;
    }
    __syncthreads();
    if (j < 100) {
        float a = fc2b[j];
        for (int i = 0; i < 100; ++i) a += sh1[i] * fc2w[i * 100 + j];
        out[(size_t)u * 100 + j] = a;
    }
}

// ---------------------------------------------------------------------------
// GRU elementwise cell update (PyTorch gate math), in-place on h.
// ---------------------------------------------------------------------------
__global__ void gru_update_kernel(const float* __restrict__ gi,
                                  const float* __restrict__ gh,
                                  float* __restrict__ h, int rows)
{
    int i = blockIdx.x * blockDim.x + threadIdx.x;
    if (i >= rows * HID) return;
    int n = i / HID, j = i - n * HID;
    const float* gir = gi + (size_t)n * G3;
    const float* ghr = gh + (size_t)n * G3;
    float ir = gir[j], iz = gir[HID + j], in_ = gir[2 * HID + j];
    float hr = ghr[j], hz = ghr[HID + j], hn_ = ghr[2 * HID + j];
    float r  = 1.0f / (1.0f + __expf(-(ir + hr)));
    float z  = 1.0f / (1.0f + __expf(-(iz + hz)));
    float nn = tanhf(in_ + r * hn_);
    float ho = h[i];
    h[i] = (1.0f - z) * nn + z * ho;
}

__global__ void fill_kernel(float* __restrict__ p, float v, int n)
{
    int i = blockIdx.x * blockDim.x + threadIdx.x;
    if (i < n) p[i] = v;
}

__global__ void copy_kernel(const float* __restrict__ s, float* __restrict__ d, int n)
{
    int i = blockIdx.x * blockDim.x + threadIdx.x;
    if (i < n) d[i] = s[i];
}

// x_input = [ h1[:4000] ; user_embedding ; h1[4000:] ]
__global__ void build_xin_kernel(const float* __restrict__ h1,
                                 const float* __restrict__ uemb,
                                 float* __restrict__ xin)
{
    int i = blockIdx.x * blockDim.x + threadIdx.x;
    if (i >= NNODES * HID) return;
    int row = i / HID, j = i - row * HID;
    float v;
    if (row < BATCH)                 v = h1[(size_t)row * HID + j];
    else if (row < BATCH + NUSERS)   v = uemb[(size_t)(row - BATCH) * HID + j];
    else                             v = h1[(size_t)(row - NUSERS) * HID + j];
    xin[i] = v;
}

__global__ void deg_count_kernel(const int* __restrict__ dst, float* __restrict__ deg)
{
    int e = blockIdx.x * blockDim.x + threadIdx.x;
    if (e < NEDGES) atomicAdd(&deg[dst[e]], 1.0f);
}

__global__ void deg_rsqrt_kernel(float* __restrict__ deg)
{
    int i = blockIdx.x * blockDim.x + threadIdx.x;
    if (i < NNODES) deg[i] = rsqrtf(deg[i]);   // deg >= 1 (self-loop)
}

// edge-parallel scatter:  acc[dst] += dinv[src]*dinv[dst] * xw[src]
__global__ void edge_scatter_kernel(const int* __restrict__ src,
                                    const int* __restrict__ dst,
                                    const float* __restrict__ dinv,
                                    const float* __restrict__ xw,
                                    float* __restrict__ acc, int F)
{
    long long i = (long long)blockIdx.x * blockDim.x + threadIdx.x;
    if (i >= (long long)NEDGES * F) return;
    int e = (int)(i / F), f = (int)(i - (long long)e * F);
    int s = src[e], d = dst[e];
    float norm = dinv[s] * dinv[d];
    atomicAdd(&acc[(size_t)d * F + f], norm * xw[(size_t)s * F + f]);
}

// self-loop contribution + bias + ELU, in-place on acc
__global__ void selfloop_bias_elu_kernel(const float* __restrict__ dinv,
                                         const float* __restrict__ xw,
                                         const float* __restrict__ bias,
                                         float* __restrict__ acc, int F)
{
    long long i = (long long)blockIdx.x * blockDim.x + threadIdx.x;
    if (i >= (long long)NNODES * F) return;
    int n = (int)(i / F), f = (int)(i - (long long)n * F);
    float di = dinv[n];
    float v = acc[i] + di * di * xw[i] + bias[f];
    acc[i] = v > 0.0f ? v : (__expf(v) - 1.0f);
}

// ---------------------------------------------------------------------------
extern "C" void kernel_launch(void* const* d_in, const int* in_sizes, int n_in,
                              void* d_out, int out_size, void* d_ws, size_t ws_size,
                              hipStream_t stream)
{
    (void)in_sizes; (void)n_in; (void)out_size; (void)ws_size;
    const float* user_feats = (const float*)d_in[0];
    const int*   tokens     = (const int*)d_in[1];
    const int*   edge       = (const int*)d_in[2];
    /* d_in[3] = indices (arange) -> batch size is static, unused */
    const float* emb        = (const float*)d_in[4];
    const float* w_ih0      = (const float*)d_in[5];
    const float* w_hh0      = (const float*)d_in[6];
    const float* b_ih0      = (const float*)d_in[7];
    const float* b_hh0      = (const float*)d_in[8];
    const float* w_ih1      = (const float*)d_in[9];
    const float* w_hh1      = (const float*)d_in[10];
    const float* b_ih1      = (const float*)d_in[11];
    const float* b_hh1      = (const float*)d_in[12];
    const float* fc1_w      = (const float*)d_in[13];
    const float* fc1_b      = (const float*)d_in[14];
    const float* fc2_w      = (const float*)d_in[15];
    const float* fc2_b      = (const float*)d_in[16];
    const float* conv1_w    = (const float*)d_in[17];
    const float* conv1_b    = (const float*)d_in[18];
    const float* conv2_w    = (const float*)d_in[19];
    const float* conv2_b    = (const float*)d_in[20];
    float* out = (float*)d_out;

    const int* e_src = edge;
    const int* e_dst = edge + NEDGES;

    // ---- workspace layout (bytes); aliased regions reused across phases ----
    char* ws = (char*)d_ws;
    float* gi   = (float*)(ws + 0);            // 40000*300*4 = 48 MB
    float* gh   = (float*)(ws + 48000000ULL);  // 48 MB
    float* h0   = (float*)(ws + 96000000ULL);  // 16 MB
    float* h1   = (float*)(ws + 112000000ULL); // 16 MB
    float* uemb = (float*)(ws + 128000000ULL); // 12 MB
    float* dinv = (float*)(ws + 140000000ULL); // 280 KB
    // post-GRU aliases (safe by phase ordering):
    float* xin  = gi;   // 28 MB node features   (gi dead after GRU)
    float* xw   = gh;   // 28 MB x @ W           (gh dead after GRU)
    float* acc1 = h0;   // 18 MB conv1 aggregate (spans h0+h1; after xin built)
    float* agg2 = gi;   // conv2 aggregate reuses xin region (xin dead then)

    const int TB = 256;
    dim3 blk128(128);

    // 1) user encoder
    user_mlp_kernel<<<NUSERS, blk128, 0, stream>>>(user_feats, fc1_w, fc1_b,
                                                   fc2_w, fc2_b, uemb);
    // 2) zero hidden states
    fill_kernel<<<(NTWEETS * HID + TB - 1) / TB, TB, 0, stream>>>(h0, 0.0f, NTWEETS * HID);
    fill_kernel<<<(NTWEETS * HID + TB - 1) / TB, TB, 0, stream>>>(h1, 0.0f, NTWEETS * HID);

    // 3) fused 2-layer GRU timestep loop
    dim3 ggru(NTWEETS / 32, 3);   // 1250 M-tiles x (3*8=24 >= 19 N-subtiles)
    int gru_blocks = (NTWEETS * HID + TB - 1) / TB;
    for (int t = 0; t < SEQ_LEN; ++t) {
        // layer 0: gi = embed(tokens[:,t]) @ w_ih0 + b_ih0 ; gh = h0 @ w_hh0 + b_hh0
        wmma_gemm_kernel<<<ggru, blk128, 0, stream>>>(nullptr, 0, tokens, t, emb,
                                                      w_ih0, b_ih0, gi,
                                                      NTWEETS, G3, EMBED, 1);
        wmma_gemm_kernel<<<ggru, blk128, 0, stream>>>(h0, HID, nullptr, 0, nullptr,
                                                      w_hh0, b_hh0, gh,
                                                      NTWEETS, G3, HID, 0);
        gru_update_kernel<<<gru_blocks, TB, 0, stream>>>(gi, gh, h0, NTWEETS);
        // layer 1: input is h0 (layer-0 output at step t)
        wmma_gemm_kernel<<<ggru, blk128, 0, stream>>>(h0, HID, nullptr, 0, nullptr,
                                                      w_ih1, b_ih1, gi,
                                                      NTWEETS, G3, HID, 0);
        wmma_gemm_kernel<<<ggru, blk128, 0, stream>>>(h1, HID, nullptr, 0, nullptr,
                                                      w_hh1, b_hh1, gh,
                                                      NTWEETS, G3, HID, 0);
        gru_update_kernel<<<gru_blocks, TB, 0, stream>>>(gi, gh, h1, NTWEETS);
    }

    // 4) assemble GCN input  [tweets[:4000]; users; tweets[4000:]]
    build_xin_kernel<<<(NNODES * HID + TB - 1) / TB, TB, 0, stream>>>(h1, uemb, xin);

    // 5) symmetric normalization: deg (with self loops) -> deg^{-1/2}
    fill_kernel<<<(NNODES + TB - 1) / TB, TB, 0, stream>>>(dinv, 1.0f, NNODES);
    deg_count_kernel<<<(NEDGES + TB - 1) / TB, TB, 0, stream>>>(e_dst, dinv);
    deg_rsqrt_kernel<<<(NNODES + TB - 1) / TB, TB, 0, stream>>>(dinv);

    // 6) conv1: xw = xin @ conv1_w  (70000x100x64), aggregate, +b, ELU
    dim3 gc1((NNODES + 31) / 32, 1);    // 4 N-subtiles (64 cols)
    wmma_gemm_kernel<<<gc1, blk128, 0, stream>>>(xin, HID, nullptr, 0, nullptr,
                                                 conv1_w, nullptr, xw,
                                                 NNODES, 64, HID, 0);
    fill_kernel<<<(NNODES * 64 + TB - 1) / TB, TB, 0, stream>>>(acc1, 0.0f, NNODES * 64);
    {
        long long n = (long long)NEDGES * 64;
        edge_scatter_kernel<<<(unsigned)((n + TB - 1) / TB), TB, 0, stream>>>(
            e_src, e_dst, dinv, xw, acc1, 64);
        long long m = (long long)NNODES * 64;
        selfloop_bias_elu_kernel<<<(unsigned)((m + TB - 1) / TB), TB, 0, stream>>>(
            dinv, xw, conv1_b, acc1, 64);
    }

    // 7) conv2: xw2 = x1 @ conv2_w (70000x64x100), aggregate, +b, ELU
    dim3 gc2((NNODES + 31) / 32, 1);    // 7 N-subtiles (100 cols)
    wmma_gemm_kernel<<<gc2, blk128, 0, stream>>>(acc1, 64, nullptr, 0, nullptr,
                                                 conv2_w, nullptr, xw,
                                                 NNODES, HID, 64, 0);
    fill_kernel<<<(NNODES * HID + TB - 1) / TB, TB, 0, stream>>>(agg2, 0.0f, NNODES * HID);
    {
        long long n = (long long)NEDGES * HID;
        edge_scatter_kernel<<<(unsigned)((n + TB - 1) / TB), TB, 0, stream>>>(
            e_src, e_dst, dinv, xw, agg2, HID);
        long long m = (long long)NNODES * HID;
        selfloop_bias_elu_kernel<<<(unsigned)((m + TB - 1) / TB), TB, 0, stream>>>(
            dinv, xw, conv2_b, agg2, HID);
    }

    // 8) output = first BATCH rows (contiguous)
    copy_kernel<<<(BATCH * HID + TB - 1) / TB, TB, 0, stream>>>(agg2, out, BATCH * HID);
}
